// GatedMemoryTitan_53455162966288
// MI455X (gfx1250) — compile-verified, matrix operand order checked by reading, not run on hardware
//
#include <hip/hip_runtime.h>
#include <hip/hip_bf16.h>

typedef unsigned short u16;
typedef __attribute__((ext_vector_type(16))) __bf16 v16bf;
typedef __attribute__((ext_vector_type(8)))  float  v8f;

#define DD 1024
#define HH 16
#define HD 64
#define PP 32
#define SS 2048
#define BB 4
#define MM (BB*SS)   // 8192 rows

union FragBF {
  v16bf v;
  uint4 q[2];
  u16   u[16];
};

__device__ __forceinline__ u16 f2bf(float f){
  unsigned x = __float_as_uint(f);
  unsigned r = (x + 0x7fffu + ((x >> 16) & 1u)) >> 16;  // RNE
  return (u16)r;
}

// async DMA: global -> LDS, 16B per lane, tracked by ASYNCcnt
__device__ __forceinline__ void async_b128(unsigned lds, const u16* gptr){
  asm volatile("global_load_async_to_lds_b128 %0, %1, off"
               :: "v"(lds), "v"((unsigned long long)(uintptr_t)gptr)
               : "memory");
}

// ---------------- f32 -> bf16 transposed weight conversion ----------------
// src f32 [R,C] row-major -> dst bf16 [C,R] (so GEMM B is always [N,K])
__global__ __launch_bounds__(256)
void k_cvt_t(const float* __restrict__ src, u16* __restrict__ dst,
             int R, int C, float scale){
  __shared__ u16 t[32][33];
  int r0 = blockIdx.y * 32, c0 = blockIdx.x * 32;
  int c = threadIdx.x & 31, rr = threadIdx.x >> 5;
  #pragma unroll
  for(int it = 0; it < 4; it++){
    int r = rr + it * 8;
    t[r][c] = f2bf(src[(long)(r0 + r) * C + c0 + c] * scale);
  }
  __syncthreads();
  int r2 = threadIdx.x & 31, cc = threadIdx.x >> 5;
  #pragma unroll
  for(int it = 0; it < 4; it++){
    int c2 = cc + it * 8;
    dst[(long)(c0 + c2) * R + r0 + r2] = t[r2][c2];
  }
}

// bf16 tile transpose: src [R,C] -> dst [C,R], batched over z
__global__ __launch_bounds__(256)
void k_tr(const u16* __restrict__ src, u16* __restrict__ dst,
          int R, int C, long sSrc, long sDst){
  __shared__ u16 t[32][33];
  src += (long)blockIdx.z * sSrc;
  dst += (long)blockIdx.z * sDst;
  int r0 = blockIdx.y * 32, c0 = blockIdx.x * 32;
  int c = threadIdx.x & 31, rr = threadIdx.x >> 5;
  #pragma unroll
  for(int it = 0; it < 4; it++){
    int r = rr + it * 8;
    t[r][c] = src[(long)(r0 + r) * C + c0 + c];
  }
  __syncthreads();
  int r2 = threadIdx.x & 31, cc = threadIdx.x >> 5;
  #pragma unroll
  for(int it = 0; it < 4; it++){
    int c2 = cc + it * 8;
    dst[(long)(c0 + c2) * R + r0 + r2] = t[r2][c2];
  }
}

// combined = [persistent ; x]  ->  bf16 [B*S, D]
__global__ void k_combined(const float* __restrict__ x, const float* __restrict__ pm,
                           u16* __restrict__ dst){
  long i = (long)blockIdx.x * blockDim.x + threadIdx.x;
  if(i >= (long)MM * DD) return;
  int d = (int)(i % DD);
  long row = i / DD;
  int s = (int)(row % SS);
  int b = (int)(row / SS);
  float v = (s < PP) ? pm[(long)s * DD + d]
                     : x[((long)b * (SS - PP) + (s - PP)) * DD + d];
  dst[i] = f2bf(v);
}

// ---------------- WMMA bf16 GEMM with async-LDS double buffering ----------------
// C[M,N] = alpha * A[M,K] @ Bt[N,K]^T + bias
// EPI: 0 = f32 out, 1 = bf16 out, 2 = gated fusion (sigmoid gate, out = g*gx + (1-g)*gy)
// 256 threads = 8 waves as 4(M) x 2(N); each wave owns a 32x32 tile (2x2 WMMAs)
// -> 128 x 64 macro tile. Tiles staged by GLOBAL_LOAD_ASYNC_TO_LDS_B128 (ASYNCcnt),
// ping-pong LDS buffers, next tile's DMA in flight underneath the WMMAs.
template<int EPI>
__global__ __launch_bounds__(256)
void k_gemm(const u16* __restrict__ A, const u16* __restrict__ Bt,
            const float* __restrict__ bias, void* __restrict__ out,
            const float* __restrict__ gx, const float* __restrict__ gy,
            int M, int N, int K, float alpha,
            long sA, long sB, long sO)
{
  A  += (long)blockIdx.z * sA;
  Bt += (long)blockIdx.z * sB;
  const int SA = 48, SB = 48;                       // half strides (96B rows, 16B aligned)
  __shared__ __align__(16) u16 As[2][128 * 48];     // A tiles, row-major [m][k]
  __shared__ __align__(16) u16 Bs[2][64 * 48];      // B tiles [n][k]
  int tid  = threadIdx.x;
  int lane = tid & 31, wid = tid >> 5;
  int wm = wid >> 1, wn = wid & 1;
  int hsel = (lane < 16) ? 0 : 8;                   // wave32 A/B fragment K-select
  int l16  = lane & 15;
  int row0 = blockIdx.y * 128;
  int col0 = blockIdx.x * 64;

  // per-lane DMA chunk assignment: 16B (8 halfs) per lane per instruction
  // A tile: 128x32 halfs = 512 chunks = 8 waves * 32 lanes * 2
  // B tile:  64x32 halfs = 256 chunks = 8 waves * 32 lanes * 1
  int a_row[2], a_k8[2];
  #pragma unroll
  for(int j = 0; j < 2; j++){
    int c = (wid * 2 + j) * 32 + lane;
    a_row[j] = c >> 2; a_k8[j] = (c & 3) * 8;
  }
  int cb = wid * 32 + lane;
  int b_row = cb >> 2, b_k8 = (cb & 3) * 8;

  const u16* aPtr[2];
  aPtr[0] = A + (long)(row0 + a_row[0]) * K + a_k8[0];
  aPtr[1] = A + (long)(row0 + a_row[1]) * K + a_k8[1];
  const u16* bPtr = Bt + (long)(col0 + b_row) * K + b_k8;
  unsigned aLds[2][2], bLds[2];
  #pragma unroll
  for(int bf = 0; bf < 2; bf++){
    aLds[bf][0] = (unsigned)(uintptr_t)&As[bf][a_row[0] * SA + a_k8[0]];
    aLds[bf][1] = (unsigned)(uintptr_t)&As[bf][a_row[1] * SA + a_k8[1]];
    bLds[bf]    = (unsigned)(uintptr_t)&Bs[bf][b_row * SB + b_k8];
  }

  v8f acc[2][2];
  v8f vzero = {};
  acc[0][0] = vzero; acc[0][1] = vzero; acc[1][0] = vzero; acc[1][1] = vzero;

  int nk = K >> 5;
  // prologue: DMA tile 0
  async_b128(aLds[0][0], aPtr[0]);
  async_b128(aLds[0][1], aPtr[1]);
  async_b128(bLds[0],    bPtr);

  for(int kt = 0; kt < nk; kt++){
    int cur = kt & 1;
    __syncthreads();                                 // all waves done reading buf (kt-1)
    if(kt + 1 < nk){                                 // DMA next tile into other buffer
      int k0 = (kt + 1) << 5;
      int nxt = cur ^ 1;
      async_b128(aLds[nxt][0], aPtr[0] + k0);
      async_b128(aLds[nxt][1], aPtr[1] + k0);
      async_b128(bLds[nxt],    bPtr + k0);
      asm volatile("s_wait_asynccnt 0x3" ::: "memory"); // tile kt retired; kt+1 in flight
    } else {
      asm volatile("s_wait_asynccnt 0x0" ::: "memory");
    }
    __syncthreads();                                 // tile kt visible to all waves
    FragBF fa[2], fb[2];
    #pragma unroll
    for(int mi = 0; mi < 2; mi++){
      int r = (wm * 32 + mi * 16 + l16) * SA;
      fa[mi].q[0] = *(const uint4*)&As[cur][r + hsel];
      fa[mi].q[1] = *(const uint4*)&As[cur][r + 16 + hsel];
    }
    #pragma unroll
    for(int ni = 0; ni < 2; ni++){
      int r = (wn * 32 + ni * 16 + l16) * SB;
      fb[ni].q[0] = *(const uint4*)&Bs[cur][r + hsel];
      fb[ni].q[1] = *(const uint4*)&Bs[cur][r + 16 + hsel];
    }
    #pragma unroll
    for(int mi = 0; mi < 2; mi++)
      #pragma unroll
      for(int ni = 0; ni < 2; ni++)
        acc[mi][ni] = __builtin_amdgcn_wmma_f32_16x16x32_bf16(
            false, fa[mi].v, false, fb[ni].v, (short)0, acc[mi][ni], false, false);
  }
  // epilogue
  #pragma unroll
  for(int ni = 0; ni < 2; ni++){
    int gn = col0 + wn * 32 + ni * 16 + l16;
    float bvv = bias ? bias[gn] : 0.0f;
    #pragma unroll
    for(int mi = 0; mi < 2; mi++){
      #pragma unroll
      for(int r = 0; r < 8; r++){
        int gm = row0 + wm * 32 + mi * 16 + r + hsel;   // C layout: M = r + (lane<16?0:8)
        long idx = (long)blockIdx.z * sO + (long)gm * N + gn;
        float val = acc[mi][ni][r] * alpha + bvv;
        if(EPI == 0)      ((float*)out)[idx] = val;
        else if(EPI == 1) ((u16*)out)[idx] = f2bf(val);
        else {
          float g = 1.0f / (1.0f + __expf(-val));
          ((float*)out)[idx] = g * gx[idx] + (1.0f - g) * gy[idx];
        }
      }
    }
  }
}

// ---------------- sliding-window flash attention (1 wave / 16-query tile) ----------------
__global__ __launch_bounds__(32)
void k_swattn(const u16* __restrict__ qb, const u16* __restrict__ kb,
              const u16* __restrict__ vb, u16* __restrict__ ob,
              const int* __restrict__ wptr){
  __shared__ __align__(16) u16 Plds[16 * 32];       // P tile, 32 keys per PV step
  int bid = blockIdx.x;
  int qt = bid & 127;
  int h  = (bid >> 7) & 15;
  int b  = bid >> 11;
  int i0 = qt * 16;
  int lane = threadIdx.x & 31;
  int hsel = (lane < 16) ? 0 : 8;
  int l16  = lane & 15;
  int W = *wptr;

  long qrow = ((long)(b * SS + i0 + l16)) * DD + h * HD;
  FragBF aq0, aq1;
  aq0.q[0] = *(const uint4*)(qb + qrow + hsel);
  aq0.q[1] = *(const uint4*)(qb + qrow + 16 + hsel);
  aq1.q[0] = *(const uint4*)(qb + qrow + 32 + hsel);
  aq1.q[1] = *(const uint4*)(qb + qrow + 48 + hsel);

  float mrun[8], lrun[8];
  v8f oacc[4];
  v8f vzero = {};
  #pragma unroll
  for(int r = 0; r < 8; r++){ mrun[r] = -1e30f; lrun[r] = 0.0f; }
  #pragma unroll
  for(int t = 0; t < 4; t++) oacc[t] = vzero;

  int ntiles = W / 16 + 1;                          // key tiles spanning [i-W, i]
  int pairs  = (ntiles + 1) >> 1;
  int start  = i0 + 16 - pairs * 32;                // lowest key (may be < 0, masked)

  for(int ps = 0; ps < pairs; ps++){
    int kb0 = start + ps * 32;
    float sc[2][8];
    #pragma unroll
    for(int t2 = 0; t2 < 2; t2++){
      int kbase = kb0 + t2 * 16;
      int rk = kbase + l16; if(rk < 0) rk = 0; if(rk >= SS) rk = SS - 1;
      long krow = ((long)(b * SS + rk)) * DD + h * HD;
      FragBF bk0, bk1;
      bk0.q[0] = *(const uint4*)(kb + krow + hsel);
      bk0.q[1] = *(const uint4*)(kb + krow + 16 + hsel);
      bk1.q[0] = *(const uint4*)(kb + krow + 32 + hsel);
      bk1.q[1] = *(const uint4*)(kb + krow + 48 + hsel);
      v8f c = vzero;
      c = __builtin_amdgcn_wmma_f32_16x16x32_bf16(false, aq0.v, false, bk0.v,
                                                  (short)0, c, false, false);
      c = __builtin_amdgcn_wmma_f32_16x16x32_bf16(false, aq1.v, false, bk1.v,
                                                  (short)0, c, false, false);
      #pragma unroll
      for(int r = 0; r < 8; r++){
        int qi = i0 + r + hsel;
        int kj = kbase + l16;
        bool ok = (kj >= 0) && (kj <= qi) && (kj >= qi - W);
        sc[t2][r] = ok ? c[r] * 0.125f : -1e30f;    // 1/sqrt(64)
      }
    }
    #pragma unroll
    for(int r = 0; r < 8; r++){
      float v = fmaxf(sc[0][r], sc[1][r]);
      v = fmaxf(v, __shfl_xor(v, 1, 32));
      v = fmaxf(v, __shfl_xor(v, 2, 32));
      v = fmaxf(v, __shfl_xor(v, 4, 32));
      v = fmaxf(v, __shfl_xor(v, 8, 32));
      float mn = fmaxf(mrun[r], v);
      float scl = __expf(mrun[r] - mn);
      lrun[r] *= scl;
      #pragma unroll
      for(int t = 0; t < 4; t++) oacc[t][r] *= scl;
      float p0 = (sc[0][r] > -9e29f) ? __expf(sc[0][r] - mn) : 0.0f;
      float p1 = (sc[1][r] > -9e29f) ? __expf(sc[1][r] - mn) : 0.0f;
      float rs = p0 + p1;
      rs += __shfl_xor(rs, 1, 32);
      rs += __shfl_xor(rs, 2, 32);
      rs += __shfl_xor(rs, 4, 32);
      rs += __shfl_xor(rs, 8, 32);
      lrun[r] += rs;
      mrun[r] = mn;
      int Mr = r + hsel;
      Plds[Mr * 32 + l16]      = f2bf(p0);
      Plds[Mr * 32 + 16 + l16] = f2bf(p1);
    }
    asm volatile("s_wait_dscnt 0" ::: "memory");    // stores -> cross-lane LDS reads
    FragBF ap;
    ap.q[0] = *(const uint4*)&Plds[l16 * 32 + hsel];
    ap.q[1] = *(const uint4*)&Plds[l16 * 32 + 16 + hsel];
    #pragma unroll
    for(int t = 0; t < 4; t++){
      FragBF bvf;
      #pragma unroll
      for(int i = 0; i < 16; i++){
        int koff = (i < 8) ? (hsel + i) : (16 + hsel + (i - 8));
        int kk = kb0 + koff; if(kk < 0) kk = 0; if(kk >= SS) kk = SS - 1;
        bvf.u[i] = vb[((long)(b * SS + kk)) * DD + h * HD + t * 16 + l16];
      }
      oacc[t] = __builtin_amdgcn_wmma_f32_16x16x32_bf16(false, ap.v, false, bvf.v,
                                                        (short)0, oacc[t], false, false);
    }
  }
  #pragma unroll
  for(int r = 0; r < 8; r++){
    float inv = (lrun[r] > 0.0f) ? 1.0f / lrun[r] : 0.0f;
    int gm = i0 + r + hsel;
    #pragma unroll
    for(int t = 0; t < 4; t++)
      ob[((long)(b * SS + gm)) * DD + h * HD + t * 16 + l16] = f2bf(oacc[t][r] * inv);
  }
}

// ---------------- row softmax (memory attention, scale pre-applied) ----------------
__global__ __launch_bounds__(256)
void k_softmax(const float* __restrict__ s, u16* __restrict__ p, int n){
  __shared__ float red[256];
  long row = blockIdx.x;
  const float* src = s + row * (long)n;
  int tid = threadIdx.x;
  float mx = -1e30f;
  for(int c = tid; c < n; c += 256) mx = fmaxf(mx, src[c]);
  red[tid] = mx; __syncthreads();
  for(int o = 128; o > 0; o >>= 1){ if(tid < o) red[tid] = fmaxf(red[tid], red[tid + o]); __syncthreads(); }
  mx = red[0]; __syncthreads();
  float sum = 0.0f;
  for(int c = tid; c < n; c += 256) sum += __expf(src[c] - mx);
  red[tid] = sum; __syncthreads();
  for(int o = 128; o > 0; o >>= 1){ if(tid < o) red[tid] += red[tid + o]; __syncthreads(); }
  float inv = 1.0f / red[0];
  u16* dst = p + row * (long)n;
  for(int c = tid; c < n; c += 256) dst[c] = f2bf(__expf(src[c] - mx) * inv);
}

// ---------------- layernorm: f32 out + bf16 into concat buffer ----------------
__global__ __launch_bounds__(256)
void k_ln(const float* __restrict__ in, const float* __restrict__ g,
          const float* __restrict__ bta, float* __restrict__ outf,
          u16* __restrict__ cat, int catStride, int colOff){
  __shared__ float red[256];
  long row = blockIdx.x;
  const float* src = in + row * (long)DD;
  int tid = threadIdx.x;
  float s = 0.0f;
  for(int c = tid; c < DD; c += 256) s += src[c];
  red[tid] = s; __syncthreads();
  for(int o = 128; o > 0; o >>= 1){ if(tid < o) red[tid] += red[tid + o]; __syncthreads(); }
  float mean = red[0] / DD; __syncthreads();
  float v = 0.0f;
  for(int c = tid; c < DD; c += 256){ float d = src[c] - mean; v += d * d; }
  red[tid] = v; __syncthreads();
  for(int o = 128; o > 0; o >>= 1){ if(tid < o) red[tid] += red[tid + o]; __syncthreads(); }
  float inv = rsqrtf(red[0] / DD + 1e-5f);
  float* of = outf + row * (long)DD;
  u16* oc = cat + row * (long)catStride + colOff;
  for(int c = tid; c < DD; c += 256){
    float y = (src[c] - mean) * inv * g[c] + bta[c];
    of[c] = y; oc[c] = f2bf(y);
  }
}

// ---------------- orchestration ----------------
extern "C" void kernel_launch(void* const* d_in, const int* in_sizes, int n_in,
                              void* d_out, int out_size, void* d_ws, size_t ws_size,
                              hipStream_t stream){
  const float* x   = (const float*)d_in[0];
  const float* pm  = (const float*)d_in[1];
  const float* Wq  = (const float*)d_in[2];  const float* bq  = (const float*)d_in[3];
  const float* Wk  = (const float*)d_in[4];  const float* bk  = (const float*)d_in[5];
  const float* Wv  = (const float*)d_in[6];  const float* bv  = (const float*)d_in[7];
  const float* Wo  = (const float*)d_in[8];  const float* bo  = (const float*)d_in[9];
  const float* mWq = (const float*)d_in[10]; const float* mbq = (const float*)d_in[11];
  const float* mWk = (const float*)d_in[12]; const float* mbk = (const float*)d_in[13];
  const float* mWv = (const float*)d_in[14]; const float* mbv = (const float*)d_in[15];
  const float* g1w = (const float*)d_in[16]; const float* g1b = (const float*)d_in[17];
  const float* g2w = (const float*)d_in[18]; const float* g2b = (const float*)d_in[19];
  const float* gW  = (const float*)d_in[20]; const float* gb  = (const float*)d_in[21];
  const int*   wsz = (const int*)d_in[22];
  (void)in_sizes; (void)n_in; (void)out_size; (void)ws_size;

  char* wsp = (char*)d_ws;
  size_t off = 0;
  auto alloc = [&](size_t bytes)->char*{
    char* p = wsp + off; off += (bytes + 255) & ~(size_t)255; return p;
  };
  u16* cmb   = (u16*)alloc((size_t)MM * DD * 2);
  u16* wqb   = (u16*)alloc((size_t)DD * DD * 2);   // transposed bf16 weights [N,K]
  u16* wkb   = (u16*)alloc((size_t)DD * DD * 2);
  u16* wvb   = (u16*)alloc((size_t)DD * DD * 2);
  u16* wob   = (u16*)alloc((size_t)DD * DD * 2);
  u16* mwqb  = (u16*)alloc((size_t)DD * DD * 2);
  u16* mwkb  = (u16*)alloc((size_t)DD * DD * 2);
  u16* mwvb  = (u16*)alloc((size_t)DD * DD * 2);
  u16* gwb   = (u16*)alloc((size_t)2 * DD * DD * 2);
  u16* qbf   = (u16*)alloc((size_t)MM * DD * 2);
  u16* kbf   = (u16*)alloc((size_t)MM * DD * 2);
  u16* vbf   = (u16*)alloc((size_t)MM * DD * 2);
  u16* atbf  = (u16*)alloc((size_t)MM * DD * 2);
  float* atp = (float*)alloc((size_t)MM * DD * 4);
  u16* mqb   = (u16*)alloc((size_t)MM * DD * 2);
  u16* mkb   = (u16*)alloc((size_t)MM * DD * 2);
  u16* mvb   = (u16*)alloc((size_t)MM * DD * 2);
  u16* mvt   = (u16*)alloc((size_t)MM * DD * 2);   // mv transposed [B, D, S]
  float* msc = (float*)alloc((size_t)BB * SS * SS * 4);
  u16* pbf   = (u16*)alloc((size_t)BB * SS * SS * 2);
  float* mof = (float*)alloc((size_t)MM * DD * 4);
  float* naf = (float*)alloc((size_t)MM * DD * 4);
  float* nmf = (float*)alloc((size_t)MM * DD * 4);
  u16* catb  = (u16*)alloc((size_t)MM * 2 * DD * 2);

  // transposed bf16 weight conversion (fold 0.1*combined of mem_state into mWk/mWv)
  dim3 gT(DD / 32, DD / 32, 1);
  k_cvt_t<<<gT, 256, 0, stream>>>(Wq,  wqb,  DD, DD, 1.0f);
  k_cvt_t<<<gT, 256, 0, stream>>>(Wk,  wkb,  DD, DD, 1.0f);
  k_cvt_t<<<gT, 256, 0, stream>>>(Wv,  wvb,  DD, DD, 1.0f);
  k_cvt_t<<<gT, 256, 0, stream>>>(Wo,  wob,  DD, DD, 1.0f);
  k_cvt_t<<<gT, 256, 0, stream>>>(mWq, mwqb, DD, DD, 1.0f);
  k_cvt_t<<<gT, 256, 0, stream>>>(mWk, mwkb, DD, DD, 0.1f);
  k_cvt_t<<<gT, 256, 0, stream>>>(mWv, mwvb, DD, DD, 0.1f);
  k_cvt_t<<<dim3(DD / 32, 2 * DD / 32, 1), 256, 0, stream>>>(gW, gwb, 2 * DD, DD, 1.0f);
  k_combined<<<dim3((MM * DD + 255) / 256), 256, 0, stream>>>(x, pm, cmb);

  dim3 gP(DD / 64, MM / 128, 1);   // [8192 x 1024] outputs
  // attention projections (bf16 out)
  k_gemm<1><<<gP,256,0,stream>>>(cmb, wqb, bq, qbf, nullptr, nullptr, MM, DD, DD, 1.0f, 0,0,0);
  k_gemm<1><<<gP,256,0,stream>>>(cmb, wkb, bk, kbf, nullptr, nullptr, MM, DD, DD, 1.0f, 0,0,0);
  k_gemm<1><<<gP,256,0,stream>>>(cmb, wvb, bv, vbf, nullptr, nullptr, MM, DD, DD, 1.0f, 0,0,0);
  // sliding-window flash attention
  k_swattn<<<dim3(BB * HH * (SS / 16)), 32, 0, stream>>>(qbf, kbf, vbf, atbf, wsz);
  // output projection (f32 out for LN)
  k_gemm<0><<<gP,256,0,stream>>>(atbf, wob, bo, atp, nullptr, nullptr, MM, DD, DD, 1.0f, 0,0,0);
  // memory projections (bf16 out)
  k_gemm<1><<<gP,256,0,stream>>>(cmb, mwqb, mbq, mqb, nullptr, nullptr, MM, DD, DD, 1.0f, 0,0,0);
  k_gemm<1><<<gP,256,0,stream>>>(cmb, mwkb, mbk, mkb, nullptr, nullptr, MM, DD, DD, 1.0f, 0,0,0);
  k_gemm<1><<<gP,256,0,stream>>>(cmb, mwvb, mbv, mvb, nullptr, nullptr, MM, DD, DD, 1.0f, 0,0,0);
  // memory attention: scores = (mq @ mk^T) / 32, batched over B (mk is already [N,K])
  dim3 gS(SS / 64, SS / 128, BB);
  k_gemm<0><<<gS,256,0,stream>>>(mqb, mkb, nullptr, msc, nullptr, nullptr,
                                 SS, SS, DD, 0.03125f,
                                 (long)SS * DD, (long)SS * DD, (long)SS * SS);
  k_softmax<<<dim3(MM), 256, 0, stream>>>(msc, pbf, SS);
  // transpose mv per batch so P@mv has B in [N,K] form
  k_tr<<<dim3(DD / 32, SS / 32, BB), 256, 0, stream>>>(mvb, mvt, SS, DD,
                                                       (long)SS * DD, (long)SS * DD);
  dim3 gO(DD / 64, SS / 128, BB);
  k_gemm<0><<<gO,256,0,stream>>>(pbf, mvt, nullptr, mof, nullptr, nullptr,
                                 SS, DD, SS, 1.0f,
                                 (long)SS * SS, (long)SS * DD, (long)SS * DD);
  // layernorms -> f32 + bf16 concat [na ; nm]
  k_ln<<<dim3(MM), 256, 0, stream>>>(atp, g1w, g1b, naf, catb, 2 * DD, 0);
  k_ln<<<dim3(MM), 256, 0, stream>>>(mof, g2w, g2b, nmf, catb, 2 * DD, DD);
  // gate GEMM with fused sigmoid + blend epilogue -> d_out (f32)
  k_gemm<2><<<gP,256,0,stream>>>(catb, gwb, gb, d_out, naf, nmf,
                                 MM, DD, 2 * DD, 1.0f, 0,0,0);
}
